// LSTM_Model_55637006352599
// MI455X (gfx1250) — compile-verified, matrix operand order checked by reading, not run on hardware
//
#include <hip/hip_runtime.h>

typedef _Float16 half8  __attribute__((ext_vector_type(8)));
typedef _Float16 half16 __attribute__((ext_vector_type(16)));
typedef float    f32x8  __attribute__((ext_vector_type(8)));

namespace lstmk {

constexpr int TSTEPS = 256;   // sequence length
constexpr int FDIM   = 64;    // input feature dim
constexpr int HDIM   = 35;    // hidden dim
constexpr int GDIM   = 140;   // 4*H
constexpr int BM     = 32;    // batch rows per workgroup
constexpr int NTHR   = 512;   // threads per workgroup
constexpr int NWAVE  = NTHR / 32;
constexpr int GSTR   = 144;   // gates row stride (f32), 9 N-tiles of 16
constexpr int KH     = 48;    // LDS K-stride for hidden-side weights (16B aligned rows)
constexpr int NTIL   = 9;     // ceil(140/16)

// ---- shared half-precision arena offsets (in halves) ----
constexpr int W_E0X = 0;                       // [144][64]  e0 Wih
constexpr int W_E0H = W_E0X + 144 * 64;        // [144][48]  e0 Whh
constexpr int W_E1X = W_E0H + 144 * KH;        // [144][48]  e1 Wih
constexpr int W_E1H = W_E1X + 144 * KH;        // [144][48]  e1 Whh
constexpr int W_D0X = W_E1H + 144 * KH;        // [144][64]  d0 Wih
constexpr int W_D0H = W_D0X + 144 * 64;        // [144][48]  d0 Whh
constexpr int W_D1X = W_D0H + 144 * KH;        // [144][48]  d1 Wih
constexpr int W_D1H = W_D1X + 144 * KH;        // [144][48]  d1 Whh
constexpr int W_FC  = W_D1H + 144 * KH;        // [64][48]   fc W
constexpr int A_X   = W_FC + 64 * KH;          // [32][64]   current cell input (f16)
constexpr int A_H0  = A_X + BM * 64;           // [32][64]   h of layer 0 (zero-padded K>=35)
constexpr int A_H1  = A_H0 + BM * 64;          // [32][64]   h of layer 1
constexpr int HTOT  = A_H1 + BM * 64;

__device__ __forceinline__ float sigf(float x) {
  return 1.0f / (1.0f + __expf(-x));
}
__device__ __forceinline__ float tanh_fast(float x) {
  return 2.0f / (1.0f + __expf(-2.0f * x)) - 1.0f;
}

// Load a 16x32 f16 fragment (A or B operand) from a row-major LDS buffer.
// Matches CDNA5 16-bit A/B-matrix VGPR layout:
//   lane<16  : row = tile*16+lane,   K = base+0..7  and base+16..23
//   lane>=16 : row = tile*16+lane-16, K = base+8..15 and base+24..31
__device__ __forceinline__ half16 ld_frag(const _Float16* buf, int stride,
                                          int tileRow, int kc, int lane) {
  const int lo = lane & 15;
  const int hi = lane >> 4;
  const _Float16* p = buf + (tileRow * 16 + lo) * stride + kc * 32 + hi * 8;
  union { half16 v; half8 h[2]; } u;
  u.h[0] = *(const half8*)(p);
  u.h[1] = *(const half8*)(p + 16);
  return u.v;
}

__device__ __forceinline__ f32x8 wmma16(half16 a, half16 b, f32x8 c) {
  // v_wmma_f32_16x16x32_f16 : D = A(16x32) x B(32x16) + C(16x16 f32)
  return __builtin_amdgcn_wmma_f32_16x16x32_f16(false, a, false, b,
                                                (short)0, c, false, false);
}

// gates[BM x 144] = xin[BM x 64] * Wx^T + hin[BM x 64] * Wh^T
// Wx stored [144][kwx] f16, Wh stored [144][48] f16 (N-major == K-contiguous,
// which is exactly the B-operand load pattern). `wave` is an SGPR value so
// this loop is scalar control flow: EXEC stays all-ones around every WMMA.
__device__ __forceinline__ void cell_gemm(const _Float16* xin, const _Float16* wx, int kwx,
                                          const _Float16* hin, const _Float16* wh,
                                          float* gates, int wave, int lane) {
  const int lo = lane & 15;
  const int hi = lane >> 4;
  const int m = wave & 1;  // M-tile fixed per wave -> A-fragments hoistable
  half16 ax0 = ld_frag(xin, 64, m, 0, lane);
  half16 ax1 = ld_frag(xin, 64, m, 1, lane);
  half16 ah0 = ld_frag(hin, 64, m, 0, lane);
  half16 ah1 = ld_frag(hin, 64, m, 1, lane);
  for (int n = wave >> 1; n < NTIL; n += NWAVE / 2) {
    f32x8 acc = {0.f, 0.f, 0.f, 0.f, 0.f, 0.f, 0.f, 0.f};
    acc = wmma16(ax0, ld_frag(wx, kwx, n, 0, lane), acc);
    acc = wmma16(ax1, ld_frag(wx, kwx, n, 1, lane), acc);
    acc = wmma16(ah0, ld_frag(wh, KH, n, 0, lane), acc);
    acc = wmma16(ah1, ld_frag(wh, KH, n, 1, lane), acc);
#pragma unroll
    for (int r = 0; r < 8; ++r) {
      // C-fragment layout: VGPR r, lanes 0-15 -> M=r, lanes 16-31 -> M=8+r
      gates[(m * 16 + hi * 8 + r) * GSTR + n * 16 + lo] = acc[r];
    }
  }
}

// LSTM pointwise update. c kept in registers (fixed thread->element map), h written
// back to its f16 LDS buffer (only cols 0..34; pad cols stay zero forever).
__device__ __forceinline__ void cell_act(const float* gates, const float* bias,
                                         float* creg, _Float16* hout, int tid) {
  for (int e = tid, k = 0; e < BM * HDIM; e += NTHR, ++k) {
    const int row = e / HDIM;
    const int u = e - row * HDIM;
    const float* gr = gates + row * GSTR + u;
    const float iv = gr[0]   + bias[u];
    const float fv = gr[35]  + bias[35 + u];
    const float gv = gr[70]  + bias[70 + u];
    const float ov = gr[105] + bias[105 + u];
    const float c = sigf(fv) * creg[k] + sigf(iv) * tanh_fast(gv);
    creg[k] = c;
    hout[row * 64 + u] = (_Float16)(sigf(ov) * tanh_fast(c));
  }
}

// f32 global weights -> f16 LDS block [rows][kw], real data [rows][kin].
__device__ __forceinline__ void wload(const float* src, _Float16* dst,
                                      int rows, int kin, int kw, int tid) {
  for (int i = tid; i < rows * kin; i += NTHR) {
    const int r = i / kin;
    const int c = i - r * kin;
    dst[r * kw + c] = (_Float16)src[i];
  }
}

__global__ __launch_bounds__(NTHR)
void lstm_ed_kernel(const float* __restrict__ batch,
                    const float* __restrict__ e0Wx, const float* __restrict__ e0Wh,
                    const float* __restrict__ e0b,
                    const float* __restrict__ e1Wx, const float* __restrict__ e1Wh,
                    const float* __restrict__ e1b,
                    const float* __restrict__ d0Wx, const float* __restrict__ d0Wh,
                    const float* __restrict__ d0b,
                    const float* __restrict__ d1Wx, const float* __restrict__ d1Wh,
                    const float* __restrict__ d1b,
                    const float* __restrict__ fcW, const float* __restrict__ fcb,
                    float* __restrict__ out) {
  __shared__ __attribute__((aligned(16))) _Float16 sA[HTOT];
  __shared__ float sGates[BM * GSTR];
  __shared__ float sBias[4 * 144 + 64];

  const int tid = threadIdx.x;
  // Force wave id into an SGPR: tile loops / fc branch become scalar branches,
  // so WMMA always executes with EXEC == all ones (ISA requirement).
  const int wave = __builtin_amdgcn_readfirstlane(tid >> 5);
  const int lane = tid & 31;
  const int b0 = blockIdx.x * BM;

  // ---- zero the whole f16 arena (weight pads + h-buffer pads must be true zeros) ----
  for (int i = tid; i < HTOT; i += NTHR) sA[i] = (_Float16)0.f;
  __syncthreads();

  // ---- stage weights f32 -> f16 LDS ----
  wload(e0Wx, sA + W_E0X, GDIM, 64, 64, tid);
  wload(e0Wh, sA + W_E0H, GDIM, HDIM, KH, tid);
  wload(e1Wx, sA + W_E1X, GDIM, HDIM, KH, tid);
  wload(e1Wh, sA + W_E1H, GDIM, HDIM, KH, tid);
  wload(d0Wx, sA + W_D0X, GDIM, 64, 64, tid);
  wload(d0Wh, sA + W_D0H, GDIM, HDIM, KH, tid);
  wload(d1Wx, sA + W_D1X, GDIM, HDIM, KH, tid);
  wload(d1Wh, sA + W_D1H, GDIM, HDIM, KH, tid);
  wload(fcW, sA + W_FC, 64, HDIM, KH, tid);
  for (int i = tid; i < GDIM; i += NTHR) {
    sBias[i] = e0b[i];
    sBias[144 + i] = e1b[i];
    sBias[288 + i] = d0b[i];
    sBias[432 + i] = d1b[i];
  }
  for (int i = tid; i < 64; i += NTHR) sBias[576 + i] = fcb[i];

  float c0[3] = {0.f, 0.f, 0.f};
  float c1[3] = {0.f, 0.f, 0.f};
  __syncthreads();

  // ================= encoder =================
  for (int t = 0; t < TSTEPS; ++t) {
    // stage x_t (coalesced streaming f32 loads -> f16 LDS; read-once => NT hint)
    for (int i = tid; i < BM * FDIM; i += NTHR) {
      const int r = i >> 6;
      const int cc = i & 63;
      const float xv =
          __builtin_nontemporal_load(&batch[(b0 + r) * (TSTEPS * FDIM) + t * FDIM + cc]);
      sA[A_X + i] = (_Float16)xv;
    }
    __syncthreads();
    cell_gemm(sA + A_X, sA + W_E0X, 64, sA + A_H0, sA + W_E0H, sGates, wave, lane);
    __syncthreads();
    cell_act(sGates, sBias + 0, c0, sA + A_H0, tid);
    __syncthreads();
    cell_gemm(sA + A_H0, sA + W_E1X, KH, sA + A_H1, sA + W_E1H, sGates, wave, lane);
    __syncthreads();
    cell_act(sGates, sBias + 144, c1, sA + A_H1, tid);
    __syncthreads();
  }

  // ================= decoder =================
  // initial decoder input = zeros; carries over encoder-final h/c in place
  for (int i = tid; i < BM * FDIM; i += NTHR) sA[A_X + i] = (_Float16)0.f;
  __syncthreads();

  for (int t = 0; t < TSTEPS; ++t) {
    cell_gemm(sA + A_X, sA + W_D0X, 64, sA + A_H0, sA + W_D0H, sGates, wave, lane);
    __syncthreads();
    cell_act(sGates, sBias + 288, c0, sA + A_H0, tid);
    __syncthreads();
    cell_gemm(sA + A_H0, sA + W_D1X, KH, sA + A_H1, sA + W_D1H, sGates, wave, lane);
    __syncthreads();
    cell_act(sGates, sBias + 432, c1, sA + A_H1, tid);
    __syncthreads();
    // fc: pred = h1 @ fcW^T + fcb ; store time-reversed (NT) + feed back as next input
    if (wave < 8) {
      const int m = wave & 1;
      const int n = wave >> 1;  // 4 N-tiles cover F=64
      f32x8 acc = {0.f, 0.f, 0.f, 0.f, 0.f, 0.f, 0.f, 0.f};
      acc = wmma16(ld_frag(sA + A_H1, 64, m, 0, lane), ld_frag(sA + W_FC, KH, n, 0, lane), acc);
      acc = wmma16(ld_frag(sA + A_H1, 64, m, 1, lane), ld_frag(sA + W_FC, KH, n, 1, lane), acc);
      const int lo = lane & 15;
      const int hi = lane >> 4;
#pragma unroll
      for (int r = 0; r < 8; ++r) {
        const int row = m * 16 + hi * 8 + r;
        const int col = n * 16 + lo;
        const float v = acc[r] + sBias[576 + col];
        __builtin_nontemporal_store(
            v, &out[(b0 + row) * (TSTEPS * FDIM) + (TSTEPS - 1 - t) * FDIM + col]);
        sA[A_X + row * 64 + col] = (_Float16)v;
      }
    }
    __syncthreads();
  }
}

}  // namespace lstmk

extern "C" void kernel_launch(void* const* d_in, const int* in_sizes, int n_in,
                              void* d_out, int out_size, void* d_ws, size_t ws_size,
                              hipStream_t stream) {
  (void)in_sizes; (void)n_in; (void)out_size; (void)d_ws; (void)ws_size;
  const float* batch = (const float*)d_in[0];
  const float* e0Wx = (const float*)d_in[1];
  const float* e0Wh = (const float*)d_in[2];
  const float* e0b  = (const float*)d_in[3];
  const float* e1Wx = (const float*)d_in[4];
  const float* e1Wh = (const float*)d_in[5];
  const float* e1b  = (const float*)d_in[6];
  const float* d0Wx = (const float*)d_in[7];
  const float* d0Wh = (const float*)d_in[8];
  const float* d0b  = (const float*)d_in[9];
  const float* d1Wx = (const float*)d_in[10];
  const float* d1Wh = (const float*)d_in[11];
  const float* d1b  = (const float*)d_in[12];
  const float* fcW  = (const float*)d_in[13];
  const float* fcb  = (const float*)d_in[14];
  float* out = (float*)d_out;

  const int B = 2048;
  dim3 grid(B / lstmk::BM);   // 64 persistent workgroups, one per WGP
  dim3 block(lstmk::NTHR);    // 16 wave32 waves
  lstmk::lstm_ed_kernel<<<grid, block, 0, stream>>>(
      batch, e0Wx, e0Wh, e0b, e1Wx, e1Wh, e1b,
      d0Wx, d0Wh, d0b, d1Wx, d1Wh, d1b, fcW, fcb, out);
}
